// MultiHeadAttention_6081673691579
// MI455X (gfx1250) — matrix operand order repaired and checked by reading, not compiled, not run on hardware
//
#include <hip/hip_runtime.h>

// ---------------- problem constants ----------------
#define Bc   2
#define Nc   2048
#define Mc   2048
#define DMc  1024
#define Hc   16
#define DHc  64
#define DOc  1024

typedef unsigned short u16;
typedef __attribute__((ext_vector_type(16))) __bf16 v16bf;
typedef __attribute__((ext_vector_type(8)))  float  v8f;
typedef __attribute__((ext_vector_type(8)))  u16    v8u16;
typedef __attribute__((ext_vector_type(4)))  int    v4i;

static __device__ __forceinline__ u16 f32_to_bf16(float f) {
    unsigned int x = __float_as_uint(f);
    x += 0x7fffu + ((x >> 16) & 1u);   // round-to-nearest-even
    return (u16)(x >> 16);
}

// A-fragment (16x32 bf16): lane holds row = lane%16, K runs [g*8,+8) and [16+g*8,+8).
static __device__ __forceinline__ v16bf load_fragA(const u16* p) {
    union { v8u16 u[2]; v16bf v; } c;
    c.u[0] = *reinterpret_cast<const v8u16*>(p);
    c.u[1] = *reinterpret_cast<const v8u16*>(p + 16);
    return c.v;
}

// B-fragment (32x16 bf16): lane holds col = lane%16, K runs [g*8,+8) and [16+g*8,+8).
static __device__ __forceinline__ v16bf load_fragB(const u16* p) {
    union { v8u16 u[2]; v16bf v; } c;
    c.u[0] = *reinterpret_cast<const v8u16*>(p);
    c.u[1] = *reinterpret_cast<const v8u16*>(p + 16);
    return c.v;
}

static __device__ __forceinline__ v8f wmma_bf16(v16bf a, v16bf b, v8f c) {
    return __builtin_amdgcn_wmma_f32_16x16x32_bf16(false, a, false, b, (short)0, c,
                                                   false, false);
}

// async 16B Global -> LDS copy (gfx1250 GLOBAL_LOAD_ASYNC_TO_LDS_B128, ASYNCcnt)
static __device__ __forceinline__ void async_cp16(const u16* gp, u16* lp) {
    typedef __attribute__((address_space(1))) v4i GV;
    typedef __attribute__((address_space(3))) v4i LV;
    __builtin_amdgcn_global_load_async_to_lds_b128((GV*)gp, (LV*)lp, 0, 0);
}
static __device__ __forceinline__ void wait_async0() {
    asm volatile("s_wait_asynccnt 0x0" ::: "memory");
}

// ---------------- stage 1: fp32 -> bf16 convert ----------------
__global__ void cvt_f32_bf16(const float* __restrict__ in, u16* __restrict__ out, int n) {
    for (int i = blockIdx.x * blockDim.x + threadIdx.x; i < n; i += gridDim.x * blockDim.x)
        out[i] = f32_to_bf16(in[i]);
}

// ---------------- stage 2: transpose + convert weights ----------------
// in: [Hn, Dn, En] fp32  ->  out: [Hn, En, Dn] bf16
__global__ void transpose_cvt(const float* __restrict__ in, u16* __restrict__ out,
                              int Hn, int Dn, int En) {
    int n = Hn * Dn * En;
    for (int i = blockIdx.x * blockDim.x + threadIdx.x; i < n; i += gridDim.x * blockDim.x) {
        int h   = i / (Dn * En);
        int rem = i - h * (Dn * En);
        int d   = rem / En;
        int e   = rem - d * En;
        out[(size_t)h * En * Dn + (size_t)e * Dn + d] = f32_to_bf16(in[i]);
    }
}

// ---------------- stage 3: q/k/v projection GEMM ----------------
// grid: (Nc/128, 3*Bc*Hc), 256 threads (8 waves as 4x2 over a 128x64 tile)
__global__ __launch_bounds__(256) void proj_qkv_kernel(
    const u16* __restrict__ qin, const u16* __restrict__ kin, const u16* __restrict__ vin,
    const u16* __restrict__ wqt, const u16* __restrict__ wkt, const u16* __restrict__ wvt,
    const float* __restrict__ bq, const float* __restrict__ bk, const float* __restrict__ bv,
    u16* __restrict__ q_p, u16* __restrict__ k_p, u16* __restrict__ vT_p) {
    int idx   = blockIdx.y;
    int which = idx >> 5;            // 0:q 1:k 2:v   (Bc*Hc == 32)
    int bh    = idx & 31;
    int h     = bh & (Hc - 1);
    int b     = bh >> 4;

    const u16*   X    = (which == 0) ? qin : (which == 1) ? kin : vin;
    const u16*   Wt   = (which == 0) ? wqt : (which == 1) ? wkt : wvt;
    const float* bias = (which == 0) ? bq  : (which == 1) ? bk  : bv;

    int wave = threadIdx.x >> 5;
    int lane = threadIdx.x & 31;
    int l16  = lane & 15;
    int g    = lane >> 4;

    int row0 = blockIdx.x * 128 + (wave >> 1) * 32;  // n tile base
    int col0 = (wave & 1) * 32;                      // e tile base (DH=64 -> 2 wave cols)

    const u16* Xb = X  + (size_t)b * Nc * DMc;
    const u16* Wh = Wt + (size_t)h * DHc * DMc;

    v8f acc[2][2] = {};
    for (int k0 = 0; k0 < DMc; k0 += 32) {
        v16bf a0 = load_fragA(Xb + (size_t)(row0 + l16)      * DMc + k0 + g * 8);
        v16bf a1 = load_fragA(Xb + (size_t)(row0 + 16 + l16) * DMc + k0 + g * 8);
        v16bf w0 = load_fragB(Wh + (size_t)(col0 + l16)      * DMc + k0 + g * 8);
        v16bf w1 = load_fragB(Wh + (size_t)(col0 + 16 + l16) * DMc + k0 + g * 8);
        acc[0][0] = wmma_bf16(a0, w0, acc[0][0]);
        acc[0][1] = wmma_bf16(a0, w1, acc[0][1]);
        acc[1][0] = wmma_bf16(a1, w0, acc[1][0]);
        acc[1][1] = wmma_bf16(a1, w1, acc[1][1]);
    }

    float scale = (which == 0) ? 0.125f : 1.0f;   // q * 1/sqrt(DH)
#pragma unroll
    for (int i = 0; i < 2; ++i) {
#pragma unroll
        for (int j = 0; j < 2; ++j) {
            int   e   = col0 + j * 16 + l16;
            float bv_ = bias[h * DHc + e];
#pragma unroll
            for (int r = 0; r < 8; ++r) {
                int   row = row0 + i * 16 + r + g * 8;
                float val = (acc[i][j][r] + bv_) * scale;
                u16   obf = f32_to_bf16(val);
                if (which == 2) {
                    vT_p[((size_t)bh * DHc + e) * Mc + row] = obf;       // [B,H,DH,M]
                } else {
                    u16* dst = (which == 0) ? q_p : k_p;                 // [B,H,N,DH]
                    dst[((size_t)bh * Nc + row) * DHc + e] = obf;
                }
            }
        }
    }
}

// ---------------- stage 4: fused flash attention ----------------
// grid: (Nc/64, Bc*Hc), 128 threads (4 waves, 16 query rows each).
// K/V tiles staged once per block into LDS via async Global->LDS copies.
__global__ __launch_bounds__(128) void attn_kernel(
    const u16* __restrict__ q_p, const u16* __restrict__ k_p, const u16* __restrict__ vT_p,
    const float* __restrict__ mask, u16* __restrict__ ctxf) {
    __shared__ __align__(16) u16 Kt[64][72];        // K tile:  [m_local][e]
    __shared__ __align__(16) u16 Vt[64][72];        // V^T tile:[e][m_local]
    __shared__ __align__(16) u16 Plds[4][16][72];   // per-wave 16x64 P tile

    int bh = blockIdx.y;
    int h  = bh & (Hc - 1);
    int b  = bh >> 4;

    int tid  = threadIdx.x;
    int wave = tid >> 5;
    int lane = tid & 31;
    int l16  = lane & 15;
    int g    = lane >> 4;
    int nw   = blockIdx.x * 64 + wave * 16;   // this wave's query-row base

    const u16* qb = q_p  + (size_t)bh * Nc  * DHc;
    const u16* kb = k_p  + (size_t)bh * Mc  * DHc;
    const u16* vb = vT_p + (size_t)bh * DHc * Mc;

    // q A-fragments, resident for the whole M loop (K = DH = 64 -> two k-steps)
    const u16* qrow = qb + (size_t)(nw + l16) * DHc;
    v16bf aq0 = load_fragA(qrow + 0  + g * 8);
    v16bf aq1 = load_fragA(qrow + 32 + g * 8);

    v8f   oacc[4] = {};
    float rmax[8], rsum[8];
#pragma unroll
    for (int r = 0; r < 8; ++r) { rmax[r] = -3.0e38f; rsum[r] = 0.0f; }

    for (int m0 = 0; m0 < Mc; m0 += 64) {
        // cooperatively stage K[m0:m0+64, :] and V^T[:, m0:m0+64] into LDS
        // 64 rows x 128B each; 128 threads x 16B chunks -> 4 chunks/thread/tile
#pragma unroll
        for (int p = 0; p < 4; ++p) {
            int idx2 = tid + p * 128;        // 0..511
            int r    = idx2 >> 3;            // row 0..63
            int c    = (idx2 & 7) * 8;       // element chunk within row
            async_cp16(kb + (size_t)(m0 + r) * DHc + c, &Kt[r][c]);
            async_cp16(vb + (size_t)r * Mc + m0 + c,    &Vt[r][c]);
        }
        wait_async0();       // this wave's copies complete
        __syncthreads();     // all waves' copies visible

        // S = q * K^T  (16 x 64 tile, 4 column sub-tiles) from LDS
        v8f s[4] = {};
#pragma unroll
        for (int t = 0; t < 4; ++t) {
            v16bf kf0 = load_fragB(&Kt[t * 16 + l16][0  + g * 8]);
            v16bf kf1 = load_fragB(&Kt[t * 16 + l16][32 + g * 8]);
            s[t] = wmma_bf16(aq0, kf0, s[t]);
            s[t] = wmma_bf16(aq1, kf1, s[t]);
        }
        // additive mask
#pragma unroll
        for (int t = 0; t < 4; ++t) {
#pragma unroll
            for (int r = 0; r < 8; ++r) {
                int row = nw + r + g * 8;
                int col = m0 + t * 16 + l16;
                s[t][r] += -1.0e10f * (1.0f - mask[(size_t)row * Mc + col]);
            }
        }
        // online softmax: row max (reduce across the 16 lanes holding a row)
#pragma unroll
        for (int r = 0; r < 8; ++r) {
            float tm = fmaxf(fmaxf(s[0][r], s[1][r]), fmaxf(s[2][r], s[3][r]));
#pragma unroll
            for (int mk = 1; mk < 16; mk <<= 1) tm = fmaxf(tm, __shfl_xor(tm, mk, 32));
            float nm = fmaxf(rmax[r], tm);
            float sc = __expf(rmax[r] - nm);
            rmax[r]  = nm;
            rsum[r] *= sc;
#pragma unroll
            for (int u = 0; u < 4; ++u) oacc[u][r] *= sc;
        }
        // P = exp(S - rowmax), accumulate row sums
#pragma unroll
        for (int t = 0; t < 4; ++t)
#pragma unroll
            for (int r = 0; r < 8; ++r) s[t][r] = __expf(s[t][r] - rmax[r]);
#pragma unroll
        for (int r = 0; r < 8; ++r) {
            float ps = s[0][r] + s[1][r] + s[2][r] + s[3][r];
#pragma unroll
            for (int mk = 1; mk < 16; mk <<= 1) ps += __shfl_xor(ps, mk, 32);
            rsum[r] += ps;
        }
        // stage P (bf16) through per-wave LDS region to reach A-fragment layout
#pragma unroll
        for (int t = 0; t < 4; ++t)
#pragma unroll
            for (int r = 0; r < 8; ++r)
                Plds[wave][r + g * 8][t * 16 + l16] = f32_to_bf16(s[t][r]);
        asm volatile("" ::: "memory");   // same-wave LDS is in-order; keep compiler honest

        v16bf ap0 = load_fragA(&Plds[wave][l16][0  + g * 8]);
        v16bf ap1 = load_fragA(&Plds[wave][l16][32 + g * 8]);
        // ctx += P * V   (V^T tile in LDS -> contiguous B-fragments)
#pragma unroll
        for (int u = 0; u < 4; ++u) {
            v16bf vf0 = load_fragB(&Vt[u * 16 + l16][0  + g * 8]);
            v16bf vf1 = load_fragB(&Vt[u * 16 + l16][32 + g * 8]);
            oacc[u] = wmma_bf16(ap0, vf0, oacc[u]);
            oacc[u] = wmma_bf16(ap1, vf1, oacc[u]);
        }
        __syncthreads();     // protect Kt/Vt before next iteration overwrites
    }

    // normalize and store ctx in the reference's flat layout c = d*H + h
#pragma unroll
    for (int u = 0; u < 4; ++u) {
        int d = u * 16 + l16;
#pragma unroll
        for (int r = 0; r < 8; ++r) {
            int   row = nw + r + g * 8;
            float val = oacc[u][r] / rsum[r];
            ctxf[((size_t)b * Nc + row) * (Hc * DHc) + d * Hc + h] = f32_to_bf16(val);
        }
    }
}

// ---------------- stage 5: output projection ----------------
// out[B*N, 1024] = ctxf(bf16) x Wo + bo ; grid ((B*N)/128, DOc/64), 256 threads
__global__ __launch_bounds__(256) void outproj_kernel(
    const u16* __restrict__ ctxf, const u16* __restrict__ wot,
    const float* __restrict__ bo, float* __restrict__ out) {
    int wave = threadIdx.x >> 5;
    int lane = threadIdx.x & 31;
    int l16  = lane & 15;
    int g    = lane >> 4;

    int row0 = blockIdx.x * 128 + (wave >> 1) * 32;
    int col0 = blockIdx.y * 64  + (wave & 1) * 32;

    v8f acc[2][2] = {};
    for (int k0 = 0; k0 < Hc * DHc; k0 += 32) {
        v16bf a0 = load_fragA(ctxf + (size_t)(row0 + l16)      * (Hc * DHc) + k0 + g * 8);
        v16bf a1 = load_fragA(ctxf + (size_t)(row0 + 16 + l16) * (Hc * DHc) + k0 + g * 8);
        v16bf w0 = load_fragB(wot  + (size_t)(col0 + l16)      * (Hc * DHc) + k0 + g * 8);
        v16bf w1 = load_fragB(wot  + (size_t)(col0 + 16 + l16) * (Hc * DHc) + k0 + g * 8);
        acc[0][0] = wmma_bf16(a0, w0, acc[0][0]);
        acc[0][1] = wmma_bf16(a0, w1, acc[0][1]);
        acc[1][0] = wmma_bf16(a1, w0, acc[1][0]);
        acc[1][1] = wmma_bf16(a1, w1, acc[1][1]);
    }
#pragma unroll
    for (int i = 0; i < 2; ++i)
#pragma unroll
        for (int j = 0; j < 2; ++j) {
            int   col = col0 + j * 16 + l16;
            float bb  = bo[col];
#pragma unroll
            for (int r = 0; r < 8; ++r) {
                int row = row0 + i * 16 + r + g * 8;
                out[(size_t)row * DOc + col] = acc[i][j][r] + bb;
            }
        }
}

// ---------------- host launcher ----------------
extern "C" void kernel_launch(void* const* d_in, const int* in_sizes, int n_in,
                              void* d_out, int out_size, void* d_ws, size_t ws_size,
                              hipStream_t stream) {
    (void)in_sizes; (void)n_in; (void)out_size; (void)ws_size;
    const float* query = (const float*)d_in[0];
    const float* key   = (const float*)d_in[1];
    const float* value = (const float*)d_in[2];
    const float* mask  = (const float*)d_in[3];
    const float* Wq    = (const float*)d_in[4];
    const float* bq    = (const float*)d_in[5];
    const float* Wk    = (const float*)d_in[6];
    const float* bk    = (const float*)d_in[7];
    const float* Wv    = (const float*)d_in[8];
    const float* bv    = (const float*)d_in[9];
    const float* Wo    = (const float*)d_in[10];
    const float* bo    = (const float*)d_in[11];
    float* out = (float*)d_out;

    char*  w   = (char*)d_ws;
    size_t off = 0;
    auto carve = [&](size_t elems) -> u16* {
        u16* p = (u16*)(w + off);
        off += (elems * sizeof(u16) + 255) & ~(size_t)255;
        return p;
    };
    u16* qbf  = carve((size_t)Bc * Nc * DMc);       // query  bf16
    u16* kbf  = carve((size_t)Bc * Mc * DMc);       // key    bf16
    u16* vbf  = carve((size_t)Bc * Mc * DMc);       // value  bf16
    u16* wqt  = carve((size_t)Hc * DHc * DMc);      // Wq^T   [H,DH,DM]
    u16* wkt  = carve((size_t)Hc * DHc * DMc);
    u16* wvt  = carve((size_t)Hc * DHc * DMc);
    u16* wot  = carve((size_t)DOc * DMc);           // Wo^T   [DO,DM]
    u16* q_p  = carve((size_t)Bc * Hc * Nc * DHc);  // scaled q  [B,H,N,DH]
    u16* k_p  = carve((size_t)Bc * Hc * Mc * DHc);  // k         [B,H,M,DH]
    u16* vT_p = carve((size_t)Bc * Hc * DHc * Mc);  // v^T       [B,H,DH,M]
    u16* ctxf = carve((size_t)Bc * Nc * Hc * DHc);  // ctx flat (d*H+h)

    cvt_f32_bf16<<<2048, 256, 0, stream>>>(query, qbf, Bc * Nc * DMc);
    cvt_f32_bf16<<<2048, 256, 0, stream>>>(key,   kbf, Bc * Mc * DMc);
    cvt_f32_bf16<<<2048, 256, 0, stream>>>(value, vbf, Bc * Mc * DMc);
    transpose_cvt<<<1024, 256, 0, stream>>>(Wq, wqt, Hc, DMc, DHc);
    transpose_cvt<<<1024, 256, 0, stream>>>(Wk, wkt, Hc, DMc, DHc);
    transpose_cvt<<<1024, 256, 0, stream>>>(Wv, wvt, Hc, DMc, DHc);
    transpose_cvt<<<1024, 256, 0, stream>>>(Wo, wot, 1, DMc, DOc);

    proj_qkv_kernel<<<dim3(Nc / 128, 3 * Bc * Hc), 256, 0, stream>>>(
        qbf, kbf, vbf, wqt, wkt, wvt, bq, bk, bv, q_p, k_p, vT_p);

    attn_kernel<<<dim3(Nc / 64, Bc * Hc), 128, 0, stream>>>(q_p, k_p, vT_p, mask, ctxf);

    outproj_kernel<<<dim3((Bc * Nc) / 128, DOc / 64), 256, 0, stream>>>(ctxf, wot, bo, out);
}
